// MultiHeadAttention_62672162783461
// MI455X (gfx1250) — matrix operand order repaired and checked by reading, not compiled, not run on hardware
//
#include <hip/hip_runtime.h>
#include <cmath>

// ---------------------------------------------------------------------------
// Problem constants (from reference): B=2, T=2048, D=1024, H=16, hd=64
// ---------------------------------------------------------------------------
#define BATCH 2
#define TDIM  2048
#define DDIM  1024
#define NHEAD 16
#define HDIM  64
#define MROWS (BATCH * TDIM)   // 4096

typedef __attribute__((ext_vector_type(16))) __bf16 v16bf;
typedef __attribute__((ext_vector_type(8)))  float  v8f;

union AFrag {           // 32 bytes: 16 bf16 elements (8 VGPRs)
    v16bf v;
    uint4 u[2];
};

// ---------------------------------------------------------------------------
// Elementwise converts / transpose (one-time, tiny vs GEMM cost)
// ---------------------------------------------------------------------------
__global__ void convert_x_kernel(const float* __restrict__ x,
                                 __bf16* __restrict__ xb, int n) {
    int i = blockIdx.x * blockDim.x + threadIdx.x;
    if (i < n) xb[i] = (__bf16)x[i];
}

// Wt[n*D + k] = bf16( W[k*D + n] )   (D x D)
__global__ void transpose_w_kernel(const float* __restrict__ W,
                                   __bf16* __restrict__ Wt) {
    int o = blockIdx.x * blockDim.x + threadIdx.x;   // o = n*D + k
    int n = o >> 10;
    int k = o & (DDIM - 1);
    Wt[o] = (__bf16)W[k * DDIM + n];
}

// ---------------------------------------------------------------------------
// Fused QKV projection: out = x @ W + b, then RoPE (q,k) and pack.
//  grid: (M/128, N/64 = heads, 3)   block: 128 threads (4 waves)
//  wave tile: 32 (M) x 64 (N)  -> 8 C tiles, 8 WMMA per K-step of 32
//  z==0 -> Q  (rope, layout (b,h,t,d))
//  z==1 -> K  (rope, layout (b,h,t,d))
//  z==2 -> V  (no rope, layout (b,h,d,t))
// ---------------------------------------------------------------------------
__global__ __launch_bounds__(128) void qkv_proj_kernel(
    const __bf16* __restrict__ xb,
    const __bf16* __restrict__ Wtq, const __bf16* __restrict__ Wtk,
    const __bf16* __restrict__ Wtv,
    const float* __restrict__ bq, const float* __restrict__ bk,
    const float* __restrict__ bv,
    const float* __restrict__ cosT, const float* __restrict__ sinT,
    __bf16* __restrict__ Qb, __bf16* __restrict__ Kb,
    __bf16* __restrict__ Vtb) {

    const int lane = threadIdx.x & 31;
    const int wave = threadIdx.x >> 5;
    const int hl   = lane >> 4;      // half-wave (0/1)
    const int r    = lane & 15;
    const int mBase = blockIdx.x * 128 + wave * 32;
    const int head  = blockIdx.y;            // N block == one 64-wide head
    const int nBase = head * HDIM;
    const int z = blockIdx.z;

    const __bf16* Wt   = (z == 0) ? Wtq : (z == 1) ? Wtk : Wtv;
    const float*  bias = (z == 0) ? bq  : (z == 1) ? bk  : bv;

    const v8f zero8 = {0.f, 0.f, 0.f, 0.f, 0.f, 0.f, 0.f, 0.f};
    v8f acc[2][4];
#pragma unroll
    for (int i = 0; i < 2; ++i)
#pragma unroll
        for (int j = 0; j < 4; ++j) acc[i][j] = zero8;

    // A: row-major x (K contiguous). Lane's A-row per M tile.
    const __bf16* aptr0 = xb + (size_t)(mBase + r) * DDIM;
    const __bf16* aptr1 = xb + (size_t)(mBase + 16 + r) * DDIM;
    // B: Wt row-major N x K (K contiguous). Lane's N col per N tile.
    const __bf16* bptr[4];
#pragma unroll
    for (int t = 0; t < 4; ++t)
        bptr[t] = Wt + (size_t)(nBase + t * 16 + r) * DDIM;

    for (int kk = 0; kk < DDIM; kk += 32) {
        AFrag a0, a1;
        const int ca = kk + hl * 8;            // A chunk0; chunk1 at +16
        a0.u[0] = *(const uint4*)(aptr0 + ca);
        a0.u[1] = *(const uint4*)(aptr0 + ca + 16);
        a1.u[0] = *(const uint4*)(aptr1 + ca);
        a1.u[1] = *(const uint4*)(aptr1 + ca + 16);

        AFrag bf[4];
        const int cb = kk + hl * 8;            // B chunk0; chunk1 at +16
#pragma unroll
        for (int t = 0; t < 4; ++t) {
            bf[t].u[0] = *(const uint4*)(bptr[t] + cb);
            bf[t].u[1] = *(const uint4*)(bptr[t] + cb + 16);
        }
#pragma unroll
        for (int t = 0; t < 4; ++t) {
            acc[0][t] = __builtin_amdgcn_wmma_f32_16x16x32_bf16(
                false, a0.v, false, bf[t].v, (short)0, acc[0][t], false, false);
            acc[1][t] = __builtin_amdgcn_wmma_f32_16x16x32_bf16(
                false, a1.v, false, bf[t].v, (short)0, acc[1][t], false, false);
        }
    }

    // ---- epilogue: bias (+ RoPE) + pack ----
    float bvals[4];
#pragma unroll
    for (int t = 0; t < 4; ++t) bvals[t] = bias[nBase + t * 16 + r];

#pragma unroll
    for (int mt = 0; mt < 2; ++mt) {
#pragma unroll
        for (int idx = 0; idx < 8; ++idx) {
            const int m  = mBase + mt * 16 + 8 * hl + idx;   // row of x
            const int bb = m >> 11;                          // batch
            const int tt = m & (TDIM - 1);                   // token
            float vals[4];
#pragma unroll
            for (int t = 0; t < 4; ++t) vals[t] = acc[mt][t][idx] + bvals[t];

            if (z < 2) {
                // RoPE: j<32 -> v*cos - v[j+32]*sin ; j>=32 -> v*cos + v[j-32]*sin
                __bf16* outp = (z == 0) ? Qb : Kb;
                float rv[4];
#pragma unroll
                for (int t = 0; t < 4; ++t) {
                    const int j = t * 16 + r;
                    const float c = cosT[tt * HDIM + j];
                    const float s = sinT[tt * HDIM + j];
                    const float partner = (t < 2) ? vals[t + 2] : vals[t - 2];
                    rv[t] = (t < 2) ? (vals[t] * c - partner * s)
                                    : (vals[t] * c + partner * s);
                }
                __bf16* dst =
                    outp + (((size_t)bb * NHEAD + head) * TDIM + tt) * HDIM;
#pragma unroll
                for (int t = 0; t < 4; ++t) dst[t * 16 + r] = (__bf16)rv[t];
            } else {
                // V: store transposed (b,h,d,t) so attention B-frag is contiguous
#pragma unroll
                for (int t = 0; t < 4; ++t) {
                    const int j = t * 16 + r;
                    Vtb[(((size_t)bb * NHEAD + head) * HDIM + j) * TDIM + tt] =
                        (__bf16)vals[t];
                }
            }
        }
    }
}

// ---------------------------------------------------------------------------
// Flash-style attention with async global->LDS staging of K/V tiles.
//  grid: (T/64, B*H)   block: 128 threads (4 waves)
//  Per 32-key block: the whole workgroup async-copies the shared
//  K tile (32x64 bf16, 4KB) and V tile (64x32 bf16, 4KB) into LDS once
//  (GLOBAL_LOAD_ASYNC_TO_LDS_B128, ASYNCcnt), then all 4 waves read their
//  WMMA fragments from LDS.
// ---------------------------------------------------------------------------
__global__ __launch_bounds__(128) void attn_kernel(
    const __bf16* __restrict__ Qb, const __bf16* __restrict__ Kb,
    const __bf16* __restrict__ Vtb, float* __restrict__ out) {

    __shared__ __attribute__((aligned(16))) __bf16 kt[32 * HDIM]; // [key][d]
    __shared__ __attribute__((aligned(16))) __bf16 vt[HDIM * 32]; // [d][key]
    __shared__ float pshare[4 * 16 * 32];    // per-wave 16x32 P tile (f32)

    const int lane = threadIdx.x & 31;
    const int wave = threadIdx.x >> 5;
    const int hl   = lane >> 4;
    const int r    = lane & 15;
    const int bh   = blockIdx.y;             // b*H + h
    const int bb   = bh >> 4;
    const int h    = bh & 15;
    const int qBase = blockIdx.x * 64 + wave * 16;

    // Q A-frags (d = 0..31, 32..63), loaded once.
    const __bf16* qrow = Qb + ((size_t)bh * TDIM + qBase + r) * HDIM;
    AFrag qa[2];
#pragma unroll
    for (int f = 0; f < 2; ++f) {
        const int d0 = f * 32 + hl * 8;
        qa[f].u[0] = *(const uint4*)(qrow + d0);
        qa[f].u[1] = *(const uint4*)(qrow + d0 + 16);
    }

    const v8f zero8 = {0.f, 0.f, 0.f, 0.f, 0.f, 0.f, 0.f, 0.f};
    v8f o[4];
#pragma unroll
    for (int t = 0; t < 4; ++t) o[t] = zero8;
    float mrow[8], lrow[8];
#pragma unroll
    for (int i = 0; i < 8; ++i) { mrow[i] = -3.0e38f; lrow[i] = 0.f; }

    const __bf16* kbase = Kb  + (size_t)bh * TDIM * HDIM;
    const __bf16* vbase = Vtb + (size_t)bh * HDIM * TDIM;
    float* psl = &pshare[wave * 16 * 32];

    for (int kb = 0; kb < TDIM; kb += 32) {
        // ---- async-copy the shared K and V tiles into LDS (whole block) ----
        {
            // K tile: 4KB contiguous at kbase + kb*64 -> kt (256 x 16B chunks)
            const __bf16* gk = kbase + (size_t)kb * HDIM;
#pragma unroll
            for (int j = 0; j < 2; ++j) {
                const int c = threadIdx.x + 128 * j;          // chunk id
                const unsigned loff =
                    (unsigned)(size_t)((const char*)kt + c * 16);
                const __bf16* g = gk + c * 8;
                asm volatile("global_load_async_to_lds_b128 %0, %1, off"
                             :: "v"(loff), "v"(g) : "memory");
            }
            // V tile: 64 rows (d) x 32 keys; row stride TDIM in global
#pragma unroll
            for (int j = 0; j < 2; ++j) {
                const int c  = threadIdx.x + 128 * j;         // 0..255
                const int d  = c >> 2;
                const int q4 = c & 3;                         // 16B chunk in row
                const unsigned loff =
                    (unsigned)(size_t)((const char*)vt + d * 64 + q4 * 16);
                const __bf16* g = vbase + (size_t)d * TDIM + kb + q4 * 8;
                asm volatile("global_load_async_to_lds_b128 %0, %1, off"
                             :: "v"(loff), "v"(g) : "memory");
            }
            asm volatile("s_wait_asynccnt 0" ::: "memory");
            __syncthreads();
        }

        // ---- scores: S(16x32) = Q(16x64) @ K_block(32x64)^T, two 16-col tiles
        v8f s[2];
        s[0] = zero8; s[1] = zero8;
#pragma unroll
        for (int n = 0; n < 2; ++n) {
            const __bf16* krow = kt + (n * 16 + r) * HDIM;
            AFrag kf0, kf1;
            kf0.u[0] = *(const uint4*)(krow + hl * 8);
            kf0.u[1] = *(const uint4*)(krow + hl * 8 + 16);
            kf1.u[0] = *(const uint4*)(krow + 32 + hl * 8);
            kf1.u[1] = *(const uint4*)(krow + 32 + hl * 8 + 16);
            s[n] = __builtin_amdgcn_wmma_f32_16x16x32_bf16(
                false, qa[0].v, false, kf0.v, (short)0, s[n], false, false);
            s[n] = __builtin_amdgcn_wmma_f32_16x16x32_bf16(
                false, qa[1].v, false, kf1.v, (short)0, s[n], false, false);
        }

        // ---- online softmax (row stats live uniformly in each 16-lane half)
        float sc[8];
#pragma unroll
        for (int idx = 0; idx < 8; ++idx) {
            float v0 = s[0][idx] * 0.125f;       // 1/sqrt(64)
            float v1 = s[1][idx] * 0.125f;
            float mx = fmaxf(v0, v1);
#pragma unroll
            for (int msk = 1; msk < 16; msk <<= 1)
                mx = fmaxf(mx, __shfl_xor(mx, msk, 32));
            const float mnew = fmaxf(mrow[idx], mx);
            sc[idx] = __expf(mrow[idx] - mnew);
            const float p0 = __expf(v0 - mnew);
            const float p1 = __expf(v1 - mnew);
            float sum = p0 + p1;
#pragma unroll
            for (int msk = 1; msk < 16; msk <<= 1)
                sum += __shfl_xor(sum, msk, 32);
            lrow[idx] = lrow[idx] * sc[idx] + sum;
            mrow[idx] = mnew;
            // stage P in wave-private LDS (C layout -> row-major 16x32)
            const int row = idx + 8 * hl;
            psl[row * 32 + r]      = p0;
            psl[row * 32 + 16 + r] = p1;
        }
        // rescale output accumulators
#pragma unroll
        for (int t = 0; t < 4; ++t)
#pragma unroll
            for (int idx = 0; idx < 8; ++idx) o[t][idx] *= sc[idx];

        asm volatile("s_wait_dscnt 0" ::: "memory");

        // ---- re-read P in A-frag layout (two 8-wide contiguous K chunks)
        AFrag pf;
        {
            const float* prow = psl + r * 32;
#pragma unroll
            for (int e = 0; e < 8; ++e)
                pf.v[e] = (__bf16)prow[hl * 8 + e];
#pragma unroll
            for (int e = 0; e < 8; ++e)
                pf.v[8 + e] = (__bf16)prow[16 + hl * 8 + e];
        }

        // ---- out(16x64) += P(16x32) @ V_block(32x64); V tile in LDS (d-major)
#pragma unroll
        for (int t = 0; t < 4; ++t) {
            const __bf16* vrow = vt + (t * 16 + r) * 32 + hl * 8;
            AFrag vf;
            vf.u[0] = *(const uint4*)(vrow);
            vf.u[1] = *(const uint4*)(vrow + 16);
            o[t] = __builtin_amdgcn_wmma_f32_16x16x32_bf16(
                false, pf.v, false, vf.v, (short)0, o[t], false, false);
        }

        // protect LDS tiles against next iteration's async overwrite (WAR)
        __syncthreads();
    }

    // ---- finalize: divide by softmax denominator, store (B,T,D) f32
#pragma unroll
    for (int t = 0; t < 4; ++t) {
#pragma unroll
        for (int idx = 0; idx < 8; ++idx) {
            const float res = o[t][idx] / lrow[idx];
            const int trow = qBase + idx + 8 * hl;
            out[((size_t)bb * TDIM + trow) * DDIM + h * HDIM + t * 16 + r] = res;
        }
    }
}

// ---------------------------------------------------------------------------
// Launch
// ---------------------------------------------------------------------------
extern "C" void kernel_launch(void* const* d_in, const int* in_sizes, int n_in,
                              void* d_out, int out_size, void* d_ws,
                              size_t ws_size, hipStream_t stream) {
    const float* x    = (const float*)d_in[0];
    const float* cosT = (const float*)d_in[1];
    const float* sinT = (const float*)d_in[2];
    const float* Wq   = (const float*)d_in[3];
    const float* bq   = (const float*)d_in[4];
    const float* Wk   = (const float*)d_in[5];
    const float* bk   = (const float*)d_in[6];
    const float* Wv   = (const float*)d_in[7];
    const float* bv   = (const float*)d_in[8];

    char* ws = (char*)d_ws;
    const size_t MB = 1024 * 1024;
    __bf16* xb  = (__bf16*)(ws);               //  8 MB : x in bf16
    __bf16* Wtq = (__bf16*)(ws +  8 * MB);     //  2 MB : Wq^T bf16
    __bf16* Wtk = (__bf16*)(ws + 10 * MB);     //  2 MB
    __bf16* Wtv = (__bf16*)(ws + 12 * MB);     //  2 MB
    __bf16* Qb  = (__bf16*)(ws + 14 * MB);     //  8 MB : (b,h,t,d)
    __bf16* Kb  = (__bf16*)(ws + 22 * MB);     //  8 MB : (b,h,t,d)
    __bf16* Vtb = (__bf16*)(ws + 30 * MB);     //  8 MB : (b,h,d,t)

    const int nx = MROWS * DDIM;               // 4,194,304
    convert_x_kernel<<<nx / 256, 256, 0, stream>>>(x, xb, nx);
    transpose_w_kernel<<<(DDIM * DDIM) / 256, 256, 0, stream>>>(Wq, Wtq);
    transpose_w_kernel<<<(DDIM * DDIM) / 256, 256, 0, stream>>>(Wk, Wtk);
    transpose_w_kernel<<<(DDIM * DDIM) / 256, 256, 0, stream>>>(Wv, Wtv);

    qkv_proj_kernel<<<dim3(MROWS / 128, NHEAD, 3), 128, 0, stream>>>(
        xb, Wtq, Wtk, Wtv, bq, bk, bv, cosT, sinT, Qb, Kb, Vtb);

    attn_kernel<<<dim3(TDIM / 64, BATCH * NHEAD), 128, 0, stream>>>(
        Qb, Kb, Vtb, (float*)d_out);
}
